// _NodeUpdateMLP_63488206569666
// MI455X (gfx1250) — matrix operand order repaired and checked by reading, not compiled, not run on hardware
//
#include <hip/hip_runtime.h>
#include <hip/hip_bf16.h>

typedef __attribute__((ext_vector_type(16))) __bf16 v16bf;
typedef __attribute__((ext_vector_type(8)))  __bf16 v8bf;
typedef __attribute__((ext_vector_type(8)))  float  v8f;

// ---------- helpers ----------
static __device__ __forceinline__ unsigned short f2bf(float f) {
    unsigned int u = __float_as_uint(f);
    u += 0x7FFFu + ((u >> 16) & 1u);   // round-to-nearest-even
    return (unsigned short)(u >> 16);
}
static __device__ __forceinline__ float bf2f(unsigned short h) {
    return __uint_as_float(((unsigned int)h) << 16);
}
static __device__ __forceinline__ v16bf load16(const unsigned short* p0,
                                               const unsigned short* p1) {
    union { v16bf v; v8bf h[2]; } u;
    u.h[0] = *(const v8bf*)p0;
    u.h[1] = *(const v8bf*)p1;
    return u.v;
}

// ---------- 0: zero f32 buffer ----------
__global__ void k_zero_f32(float* p, long long n) {
    long long i = (long long)blockIdx.x * blockDim.x + threadIdx.x;
    if (i < n) p[i] = 0.f;
}

// ---------- 1: scatter-add edge features into agg ----------
__global__ void k_scatter(const float4* __restrict__ ef, const int* __restrict__ dst,
                          float* __restrict__ agg, int E) {
    long long idx = (long long)blockIdx.x * blockDim.x + threadIdx.x;
    long long total = (long long)E * 64;          // 256 floats = 64 float4 per edge
    if (idx >= total) return;
    int e = (int)(idx >> 6);
    int c = (int)(idx & 63);
    float4 v = ef[idx];
    float* p = agg + (size_t)dst[e] * 256 + c * 4;
    atomicAdd(p + 0, v.x);
    atomicAdd(p + 1, v.y);
    atomicAdd(p + 2, v.z);
    atomicAdd(p + 3, v.w);
}

// ---------- 2: concat [node | agg] -> bf16 X1 [Mpad][512], zero pad rows ----------
__global__ void k_concat(const float* __restrict__ nf, const float* __restrict__ agg,
                         unsigned short* __restrict__ X1, int N, long long total) {
    long long idx = (long long)blockIdx.x * blockDim.x + threadIdx.x;
    if (idx >= total) return;
    int row = (int)(idx >> 9);
    int col = (int)(idx & 511);
    float v = 0.f;
    if (row < N)
        v = (col < 256) ? nf[(size_t)row * 256 + col]
                        : agg[(size_t)row * 256 + (col - 256)];
    X1[idx] = f2bf(v);
}

// ---------- 3: W [K][Nd] f32 -> WT [Nd][K] bf16 ----------
__global__ void k_wt(const float* __restrict__ W, unsigned short* __restrict__ WT,
                     int Kd, int Nd) {
    long long idx = (long long)blockIdx.x * blockDim.x + threadIdx.x;
    if (idx >= (long long)Kd * Nd) return;
    int k = (int)(idx / Nd);
    int n = (int)(idx % Nd);
    WT[(size_t)n * Kd + k] = f2bf(W[idx]);
}

// ---------- 4/5: bf16 WMMA GEMM: D = act(A[Mpad][K] * BT[N][K]^T + bias), bf16 out ----------
// Block: 8 waves, each wave owns a 32x128 output tile -> block tile 256x128.
template<int K, bool RELU>
__global__ __launch_bounds__(256)
void k_gemm(const unsigned short* __restrict__ A, const unsigned short* __restrict__ BT,
            const float* __restrict__ bias, unsigned short* __restrict__ D) {
    const int lane  = threadIdx.x & 31;
    const int wave  = threadIdx.x >> 5;
    const int half  = lane >> 4;       // 0: lanes 0-15, 1: lanes 16-31
    const int l16   = lane & 15;
    const int rbase = blockIdx.x * 256 + wave * 32;
    const int cbase = blockIdx.y * 128;

    v8f c[2][8];
#pragma unroll
    for (int s = 0; s < 2; ++s)
#pragma unroll
        for (int n = 0; n < 8; ++n)
            c[s][n] = {};

    const unsigned short* arow0 = A + (size_t)(rbase + l16) * K;
    const unsigned short* arow1 = arow0 + (size_t)16 * K;

    for (int kb = 0; kb < K; kb += 32) {
        // A fragment per ISA 16-bit A layout:
        //   half 0: K {kb..kb+7, kb+16..kb+23}; half 1: {kb+8..kb+15, kb+24..kb+31}
        v16bf a0 = load16(arow0 + kb + half * 8, arow0 + kb + 16 + half * 8);
        v16bf a1 = load16(arow1 + kb + half * 8, arow1 + kb + 16 + half * 8);
#pragma unroll
        for (int n = 0; n < 8; ++n) {
            // B fragment: lane = column, half 0: K {kb..kb+7, kb+16..kb+23}; half 1: {kb+8..kb+15, kb+24..kb+31}
            const unsigned short* bcol =
                BT + (size_t)(cbase + n * 16 + l16) * K + kb + half * 8;
            v16bf b = load16(bcol, bcol + 16);
            c[0][n] = __builtin_amdgcn_wmma_f32_16x16x32_bf16(
                false, a0, false, b, (short)0, c[0][n], false, false);
            c[1][n] = __builtin_amdgcn_wmma_f32_16x16x32_bf16(
                false, a1, false, b, (short)0, c[1][n], false, false);
        }
    }

    // Epilogue: C VGPR i holds row i (lanes 0-15) / row 8+i (lanes 16-31), col = l16
#pragma unroll
    for (int s = 0; s < 2; ++s) {
#pragma unroll
        for (int n = 0; n < 8; ++n) {
            int col = cbase + n * 16 + l16;
            float bv = bias[col];
            int rowoff = rbase + s * 16 + half * 8;
#pragma unroll
            for (int i = 0; i < 8; ++i) {
                float v = c[s][n][i] + bv;
                if (RELU) v = v > 0.f ? v : 0.f;
                D[(size_t)(rowoff + i) * 1024 + col] = f2bf(v);
            }
        }
    }
}

// ---------- 6: LayerNorm over 1024 (bf16 in, f32 out) ----------
__global__ __launch_bounds__(256)
void k_ln(const unsigned short* __restrict__ Y, const float* __restrict__ gamma,
          const float* __restrict__ beta, float* __restrict__ out) {
    int row = blockIdx.x;
    const unsigned short* yr = Y + (size_t)row * 1024;
    int base = threadIdx.x * 4;
    float x[4], s = 0.f, s2 = 0.f;
#pragma unroll
    for (int i = 0; i < 4; ++i) {
        x[i] = bf2f(yr[base + i]);
        s += x[i];
        s2 += x[i] * x[i];
    }
    __shared__ float sh1[256], sh2[256];
    sh1[threadIdx.x] = s; sh2[threadIdx.x] = s2;
    __syncthreads();
    for (int off = 128; off > 0; off >>= 1) {
        if (threadIdx.x < off) {
            sh1[threadIdx.x] += sh1[threadIdx.x + off];
            sh2[threadIdx.x] += sh2[threadIdx.x + off];
        }
        __syncthreads();
    }
    float mean = sh1[0] * (1.f / 1024.f);
    float var  = sh2[0] * (1.f / 1024.f) - mean * mean;
    float rs   = rsqrtf(var + 1e-5f);
#pragma unroll
    for (int i = 0; i < 4; ++i) {
        int col = base + i;
        out[(size_t)row * 1024 + col] = (x[i] - mean) * rs * gamma[col] + beta[col];
    }
}

// ---------- launch ----------
extern "C" void kernel_launch(void* const* d_in, const int* in_sizes, int n_in,
                              void* d_out, int out_size, void* d_ws, size_t ws_size,
                              hipStream_t stream) {
    const float* nf  = (const float*)d_in[0];
    const float* ef  = (const float*)d_in[1];
    const int*  eix  = (const int*)d_in[2];   // [2][E]; row 1 = dst
    const float* W1  = (const float*)d_in[4];
    const float* b1  = (const float*)d_in[5];
    const float* W2  = (const float*)d_in[6];
    const float* b2  = (const float*)d_in[7];
    const float* gam = (const float*)d_in[8];
    const float* bet = (const float*)d_in[9];

    const int N    = in_sizes[0] / 256;
    const int E    = in_sizes[1] / 256;
    const int Mpad = ((N + 255) / 256) * 256;

    // Workspace layout with liveness overlap:
    //   [0 .. )      agg f32 (dead after concat) then X1 bf16; later reused by Ybf
    //   [hOff .. )   H bf16  (GEMM1 out / GEMM2 in)
    //   [wt1 .. )    WT1 bf16, WT2 bf16
    char* ws = (char*)d_ws;
    const size_t ALN = 255;
    size_t aggB   = (size_t)N * 256 * 4;
    size_t x1Off  = (aggB + ALN) & ~ALN;
    size_t x1B    = (size_t)Mpad * 512 * 2;
    size_t yB     = (size_t)Mpad * 1024 * 2;
    size_t r0End  = x1Off + x1B; if (yB > r0End) r0End = yB;
    size_t hOff   = (r0End + ALN) & ~ALN;
    size_t hB     = (size_t)Mpad * 1024 * 2;
    size_t wt1Off = (hOff + hB + ALN) & ~ALN;
    size_t wt2Off = (wt1Off + (size_t)512 * 1024 * 2 + ALN) & ~ALN;

    float*          agg = (float*)(ws);
    unsigned short* X1  = (unsigned short*)(ws + x1Off);
    unsigned short* Ybf = (unsigned short*)(ws);          // reuses agg+X1 region
    unsigned short* H   = (unsigned short*)(ws + hOff);
    unsigned short* WT1 = (unsigned short*)(ws + wt1Off);
    unsigned short* WT2 = (unsigned short*)(ws + wt2Off);

    {
        long long n = (long long)N * 256;
        k_zero_f32<<<(unsigned)((n + 255) / 256), 256, 0, stream>>>(agg, n);
    }
    {
        long long t = (long long)E * 64;
        k_scatter<<<(unsigned)((t + 255) / 256), 256, 0, stream>>>(
            (const float4*)ef, eix + E, agg, E);
    }
    {
        long long t = (long long)Mpad * 512;
        k_concat<<<(unsigned)((t + 255) / 256), 256, 0, stream>>>(nf, agg, X1, N, t);
    }
    k_wt<<<(512 * 1024 + 255) / 256, 256, 0, stream>>>(W1, WT1, 512, 1024);
    k_wt<<<(1024 * 1024 + 255) / 256, 256, 0, stream>>>(W2, WT2, 1024, 1024);

    dim3 g1(Mpad / 256, 1024 / 128);
    k_gemm<512, true><<<g1, 256, 0, stream>>>(X1, WT1, b1, H);

    dim3 g2(Mpad / 256, 1024 / 128);
    k_gemm<1024, false><<<g2, 256, 0, stream>>>(H, WT2, b2, Ybf);

    k_ln<<<N, 256, 0, stream>>>(Ybf, gam, bet, (float*)d_out);
}
